// WindowAttention_74191265071552
// MI455X (gfx1250) — matrix operand order repaired and checked by reading, not compiled, not run on hardware
//
#include <hip/hip_runtime.h>
#include <hip/hip_bf16.h>

typedef __attribute__((ext_vector_type(16))) __bf16 v16bf;
typedef __attribute__((ext_vector_type(8)))  __bf16 v8bf;
typedef __attribute__((ext_vector_type(4)))  __bf16 v4bf;
typedef __attribute__((ext_vector_type(8)))  float  v8f;
typedef __attribute__((ext_vector_type(4)))  float  v4f;
typedef __attribute__((ext_vector_type(4)))  int    v4i;

#define WMMA_BF16(A, B, C) \
    __builtin_amdgcn_wmma_f32_16x16x32_bf16(false, (A), false, (B), (short)0, (C), false, false)

// ---- CDNA5 async global->LDS copy (16B per lane), guarded for portability ----
#if __has_builtin(__builtin_amdgcn_global_load_async_to_lds_b128)
#define USE_ASYNC_LDS 1
#else
#define USE_ASYNC_LDS 0
#endif

#if __has_builtin(__builtin_amdgcn_s_wait_asynccnt)
#define WAIT_ASYNC() __builtin_amdgcn_s_wait_asynccnt(0)
#else
#define WAIT_ASYNC() asm volatile("s_wait_asynccnt 0x0" ::: "memory")
#endif

#if USE_ASYNC_LDS
typedef __attribute__((address_space(1))) v4i as1_v4i;
typedef __attribute__((address_space(3))) v4i as3_v4i;
static __device__ __forceinline__ void async_cp16(void* lds, const void* glob) {
    __builtin_amdgcn_global_load_async_to_lds_b128(
        (as1_v4i*)(uintptr_t)glob,
        (as3_v4i*)(uint32_t)(uintptr_t)lds,
        0, 0);
}
#endif

// Build a 16xbf16 WMMA operand from two contiguous 16-byte LDS chunks
// (lowers to 2x ds_load_b128).
static __device__ __forceinline__ v16bf frag2(const __bf16* p0, const __bf16* p1) {
    v8bf lo = *(const v8bf*)p0;
    v8bf hi = *(const v8bf*)p1;
    return __builtin_shufflevector(lo, hi, 0,1,2,3,4,5,6,7,8,9,10,11,12,13,14,15);
}

// ---------------------------------------------------------------------------
// Kernel 1: QKV projection.  x[100352,384] @ qkv_w[384,1152] + qkv_b
// -> scatter to q/k/v workspaces [B=2048, H=12, L=49, D=32] bf16,
//    softmax scale folded into q.  64x64 block tile, 4 waves.
// ---------------------------------------------------------------------------
__global__ __launch_bounds__(128) void qkv_gemm_kernel(
    const float* __restrict__ x,
    const float* __restrict__ qkv_w,
    const float* __restrict__ qkv_b,
    __bf16* __restrict__ qws, __bf16* __restrict__ kws, __bf16* __restrict__ vws)
{
    __shared__ __bf16 As[64][40];        // 64 rows x 32 K (80B stride, 16B-aligned)
    __shared__ __bf16 Bf[4][32][24];     // B in fragment order: [ntile][lane][e]

    const int tid  = threadIdx.x;
    const int wave = tid >> 5;
    const int lane = tid & 31;
    const int hl   = lane >> 4;
    const int rsub = lane & 15;
    const int row0 = blockIdx.y * 64;
    const int col0 = blockIdx.x * 64;

    const int ar = tid >> 1;             // A staging: row
    const int ac = (tid & 1) * 16;       // A staging: col base (16 floats)
    const int bnt = tid >> 5;            // B staging: fragment tile
    const int bln = tid & 31;            //            lane slot
    const int bg  = bln >> 4;
    const int bn  = col0 + bnt * 16 + (bln & 15);

    v8f acc[4] = {};

    for (int k0 = 0; k0 < 384; k0 += 32) {
        // ---- stage A: 4x float4 loads -> packed bf16 b64 stores ----
        const float* xs = &x[(size_t)(row0 + ar) * 384 + k0 + ac];
        if (k0 < 352) __builtin_prefetch(xs + 32, 0, 1);
        #pragma unroll
        for (int j = 0; j < 4; j++) {
            v4f t = *(const v4f*)(xs + 4 * j);
            v4bf o;
            o[0] = (__bf16)t[0]; o[1] = (__bf16)t[1];
            o[2] = (__bf16)t[2]; o[3] = (__bf16)t[3];
            *(v4bf*)&As[ar][ac + 4 * j] = o;
        }
        // ---- stage B directly in fragment order (2x b128 stores) ----
        {
            const float* wp = &qkv_w[(size_t)(k0 + 16 * bg) * 1152 + bn];
            v8bf blo, bhi;
            #pragma unroll
            for (int e = 0; e < 8; e++) blo[e] = (__bf16)wp[(size_t)e * 1152];
            #pragma unroll
            for (int e = 0; e < 8; e++) bhi[e] = (__bf16)wp[(size_t)(e + 8) * 1152];
            *(v8bf*)&Bf[bnt][bln][0] = blo;
            *(v8bf*)&Bf[bnt][bln][8] = bhi;
            if (k0 < 352) __builtin_prefetch(wp + (size_t)32 * 1152, 0, 1);
        }
        __syncthreads();

        v16bf a = frag2(&As[wave * 16 + rsub][16 * hl],
                        &As[wave * 16 + rsub][16 * hl + 8]);
        #pragma unroll
        for (int nt = 0; nt < 4; nt++) {
            v16bf bfr = frag2(&Bf[nt][lane][0], &Bf[nt][lane][8]);
            acc[nt] = WMMA_BF16(a, bfr, acc[nt]);
        }
        __syncthreads();
    }

    // ---- epilogue: section (q/k/v) is block-uniform since 384 % 64 == 0 ----
    const int sSec = col0 / 384;
    __bf16* outp = (sSec == 0) ? qws : (sSec == 1) ? kws : vws;
    const float mul = (sSec == 0) ? 0.17677669529663687f : 1.f;   // 32^-0.5 on q

    float  bias[4];
    size_t noff[4];
    #pragma unroll
    for (int nt = 0; nt < 4; nt++) {
        int col = col0 + nt * 16 + rsub;
        int rem = col - sSec * 384;
        bias[nt] = qkv_b[col];
        noff[nt] = (size_t)(rem >> 5) * 49 * 32 + (rem & 31);  // h*L*D + d
    }
    #pragma unroll
    for (int r = 0; r < 8; r++) {
        int m  = row0 + wave * 16 + r + 8 * hl;
        int bi = m / 49, l = m - bi * 49;
        size_t rbase = (size_t)bi * 12 * 49 * 32 + (size_t)l * 32;
        #pragma unroll
        for (int nt = 0; nt < 4; nt++)
            outp[rbase + noff[nt]] = (__bf16)((acc[nt][r] + bias[nt]) * mul);
    }
}

// ---------------------------------------------------------------------------
// Kernel 2: per-(window,head) attention.  One block per (b,h), 4 waves.
// Q/K staged via async global->LDS DMA; V staged transposed; S stored
// transposed so accumulator spills are vector stores.
// ---------------------------------------------------------------------------
__global__ __launch_bounds__(128) void attn_kernel(
    const __bf16* __restrict__ qws, const __bf16* __restrict__ kws,
    const __bf16* __restrict__ vws,
    const int*   __restrict__ rel_index,
    const float* __restrict__ rel_bias_table,
    const float* __restrict__ attn_mask,
    __bf16* __restrict__ attn_out)
{
    __shared__ __bf16 Qs[64][40];   // row-major q (rows 49..63 zero)
    __shared__ __bf16 Ks[64][40];   // row-major k
    __shared__ __bf16 Vt[32][72];   // transposed v: Vt[d][l]
    __shared__ float  St[64][68];   // transposed scores: St[n][m]
    __shared__ __bf16 Ps[64][72];   // probs, row-major

    const int tid  = threadIdx.x;
    const int wave = tid >> 5;
    const int lane = tid & 31;
    const int hl   = lane >> 4;
    const int rsub = lane & 15;
    const int bh   = blockIdx.x;
    const int b    = bh / 12;
    const int h    = bh - b * 12;
    const int wdw  = b & 63;

    const size_t base = (size_t)bh * 49 * 32;

    // ---- stage Q,K: async DMA for valid rows, zero-fill pad rows ----
    #pragma unroll
    for (int ii = 0; ii < 2; ii++) {
        int ch = tid + ii * 128;                  // 256 chunks of 8 bf16
        int r = ch >> 2, c0 = (ch & 3) * 8;
#if USE_ASYNC_LDS
        if (r < 49) {
            async_cp16(&Qs[r][c0], &qws[base + r * 32 + c0]);
            async_cp16(&Ks[r][c0], &kws[base + r * 32 + c0]);
        } else {
            v8bf z = {};
            *(v8bf*)&Qs[r][c0] = z;
            *(v8bf*)&Ks[r][c0] = z;
        }
#else
        v8bf qv = {}, kv = {};
        if (r < 49) {
            qv = *(const v8bf*)&qws[base + r * 32 + c0];
            kv = *(const v8bf*)&kws[base + r * 32 + c0];
        }
        *(v8bf*)&Qs[r][c0] = qv;
        *(v8bf*)&Ks[r][c0] = kv;
#endif
    }
    // ---- stage V transposed (gather, packed 16B LDS stores) ----
    {
        int d = tid >> 2, l0 = (tid & 3) * 16;
        v8bf p0, p1;
        #pragma unroll
        for (int j = 0; j < 8; j++) {
            int l = l0 + j;
            p0[j] = (l < 49) ? vws[base + l * 32 + d] : (__bf16)0.f;
        }
        #pragma unroll
        for (int j = 0; j < 8; j++) {
            int l = l0 + 8 + j;
            p1[j] = (l < 49) ? vws[base + l * 32 + d] : (__bf16)0.f;
        }
        *(v8bf*)&Vt[d][l0]     = p0;
        *(v8bf*)&Vt[d][l0 + 8] = p1;
    }
#if USE_ASYNC_LDS
    WAIT_ASYNC();
#endif
    __syncthreads();

    const int mrow  = wave * 16 + rsub;
    const int mbase = wave * 16 + 8 * hl;

    // ---- S = Q K^T (+bias +mask), stored transposed ----
    {
        v16bf aq = frag2(&Qs[mrow][16 * hl], &Qs[mrow][16 * hl + 8]);
        #pragma unroll
        for (int nt = 0; nt < 4; nt++) {
            v16bf bk = frag2(&Ks[nt * 16 + rsub][16 * hl],
                             &Ks[nt * 16 + rsub][16 * hl + 8]);
            v8f acc = {};
            acc = WMMA_BF16(aq, bk, acc);
            int n = nt * 16 + rsub;
            v8f vals;
            #pragma unroll
            for (int r = 0; r < 8; r++) {
                int m = mbase + r;
                float v = acc[r];
                if (m < 49 && n < 49)
                    v += rel_bias_table[rel_index[m * 49 + n] * 12 + h]
                       + attn_mask[(wdw * 49 + m) * 49 + n];
                vals[r] = v;
            }
            *(v8f*)&St[n][mbase] = vals;    // 2x ds_store_b128
        }
    }
    __syncthreads();

    // ---- row softmax -> P: single exp pass + vector rescale ----
    if (tid < 64) {
        int row = tid;
        if (row < 49) {
            float mx = -1e30f;
            for (int n = 0; n < 49; n++) mx = fmaxf(mx, St[n][row]);
            float sum = 0.f;
            for (int n0 = 0; n0 < 64; n0 += 8) {
                v8bf pk;
                #pragma unroll
                for (int j = 0; j < 8; j++) {
                    int n = n0 + j;
                    float e = (n < 49) ? __expf(St[n][row] - mx) : 0.f;
                    sum += e;
                    pk[j] = (__bf16)e;
                }
                *(v8bf*)&Ps[row][n0] = pk;
            }
            float inv = 1.f / sum;
            for (int n0 = 0; n0 < 64; n0 += 8) {
                v8bf pk = *(v8bf*)&Ps[row][n0];
                #pragma unroll
                for (int j = 0; j < 8; j++)
                    pk[j] = (__bf16)((float)pk[j] * inv);
                *(v8bf*)&Ps[row][n0] = pk;
            }
        } else {
            v8bf z = {};
            for (int n0 = 0; n0 < 64; n0 += 8) *(v8bf*)&Ps[row][n0] = z;
        }
    }
    __syncthreads();

    // ---- O = P @ V ----
    #pragma unroll
    for (int nt = 0; nt < 2; nt++) {
        v8f acc = {};
        #pragma unroll
        for (int kt = 0; kt < 2; kt++) {
            v16bf ap = frag2(&Ps[mrow][kt * 32 + 16 * hl],
                             &Ps[mrow][kt * 32 + 16 * hl + 8]);
            v16bf bv = frag2(&Vt[nt * 16 + rsub][kt * 32 + 16 * hl],
                             &Vt[nt * 16 + rsub][kt * 32 + 16 * hl + 8]);
            acc = WMMA_BF16(ap, bv, acc);
        }
        #pragma unroll
        for (int r = 0; r < 8; r++) {
            int m = mbase + r;
            int d = nt * 16 + rsub;
            if (m < 49)
                attn_out[((size_t)b * 49 + m) * 384 + h * 32 + d] = (__bf16)acc[r];
        }
    }
}

// ---------------------------------------------------------------------------
// Kernel 3: output projection.  A[100352,384](bf16) @ proj_w[384,384] + proj_b
// -> d_out fp32.  A tile staged with async global->LDS DMA.
// ---------------------------------------------------------------------------
__global__ __launch_bounds__(128) void proj_gemm_kernel(
    const __bf16* __restrict__ A,
    const float*  __restrict__ proj_w,
    const float*  __restrict__ proj_b,
    float* __restrict__ out)
{
    __shared__ __bf16 As[64][40];
    __shared__ __bf16 Bf[4][32][24];

    const int tid  = threadIdx.x;
    const int wave = tid >> 5;
    const int lane = tid & 31;
    const int hl   = lane >> 4;
    const int rsub = lane & 15;
    const int row0 = blockIdx.y * 64;
    const int col0 = blockIdx.x * 64;

    const int bnt = tid >> 5;
    const int bln = tid & 31;
    const int bg  = bln >> 4;
    const int bn  = col0 + bnt * 16 + (bln & 15);

    v8f acc[4] = {};

    for (int k0 = 0; k0 < 384; k0 += 32) {
        // ---- stage A (bf16 copy): async DMA, 2x 16B per thread ----
        #pragma unroll
        for (int ii = 0; ii < 2; ii++) {
            int ch = tid * 2 + ii;                 // 256 chunks of 8 bf16
            int r = ch >> 2, c0 = (ch & 3) * 8;
            const __bf16* src = &A[(size_t)(row0 + r) * 384 + k0 + c0];
#if USE_ASYNC_LDS
            async_cp16(&As[r][c0], src);
#else
            *(v8bf*)&As[r][c0] = *(const v8bf*)src;
#endif
        }
        // ---- stage B in fragment order ----
        {
            const float* wp = &proj_w[(size_t)(k0 + 16 * bg) * 384 + bn];
            v8bf blo, bhi;
            #pragma unroll
            for (int e = 0; e < 8; e++) blo[e] = (__bf16)wp[(size_t)e * 384];
            #pragma unroll
            for (int e = 0; e < 8; e++) bhi[e] = (__bf16)wp[(size_t)(e + 8) * 384];
            *(v8bf*)&Bf[bnt][bln][0] = blo;
            *(v8bf*)&Bf[bnt][bln][8] = bhi;
            if (k0 < 352) __builtin_prefetch(wp + (size_t)32 * 384, 0, 1);
        }
#if USE_ASYNC_LDS
        WAIT_ASYNC();
#endif
        __syncthreads();

        v16bf a = frag2(&As[wave * 16 + rsub][16 * hl],
                        &As[wave * 16 + rsub][16 * hl + 8]);
        #pragma unroll
        for (int nt = 0; nt < 4; nt++) {
            v16bf bfr = frag2(&Bf[nt][lane][0], &Bf[nt][lane][8]);
            acc[nt] = WMMA_BF16(a, bfr, acc[nt]);
        }
        __syncthreads();
    }

    #pragma unroll
    for (int nt = 0; nt < 4; nt++) {
        #pragma unroll
        for (int r = 0; r < 8; r++) {
            int m   = row0 + wave * 16 + r + 8 * hl;
            int col = col0 + nt * 16 + rsub;
            out[(size_t)m * 384 + col] = acc[nt][r] + proj_b[col];
        }
    }
}

// ---------------------------------------------------------------------------
extern "C" void kernel_launch(void* const* d_in, const int* in_sizes, int n_in,
                              void* d_out, int out_size, void* d_ws, size_t ws_size,
                              hipStream_t stream)
{
    (void)in_sizes; (void)n_in; (void)out_size; (void)ws_size;

    const float* x              = (const float*)d_in[0];
    const int*   rel_index      = (const int*)  d_in[1];
    const float* attn_mask      = (const float*)d_in[2];
    const float* qkv_w          = (const float*)d_in[3];
    const float* qkv_b          = (const float*)d_in[4];
    const float* rel_bias_table = (const float*)d_in[5];
    const float* proj_w         = (const float*)d_in[6];
    const float* proj_b         = (const float*)d_in[7];

    const size_t qkv_elems = (size_t)2048 * 12 * 49 * 32;
    __bf16* qws = (__bf16*)d_ws;
    __bf16* kws = qws + qkv_elems;
    __bf16* vws = kws + qkv_elems;
    __bf16* aws = vws + qkv_elems;   // attention output [100352, 384] bf16

    dim3 g1(18, 1568);   // 1152/64 x 100352/64
    qkv_gemm_kernel<<<g1, 128, 0, stream>>>(x, qkv_w, qkv_b, qws, kws, vws);

    attn_kernel<<<2048 * 12, 128, 0, stream>>>(qws, kws, vws, rel_index,
                                               rel_bias_table, attn_mask, aws);

    dim3 g3(6, 1568);    // 384/64 x 100352/64
    proj_gemm_kernel<<<g3, 128, 0, stream>>>(aws, proj_w, proj_b, (float*)d_out);
}
